// TransformerBlock_82583631167782
// MI455X (gfx1250) — matrix operand order repaired and checked by reading, not compiled, not run on hardware
//
#include <hip/hip_runtime.h>

#define NTOK 16384
#define HIDD 768
#define INTR 3072
#define NH   12
#define DHD  64
#define WIN  256
#define MIDC (3*HIDD + INTR)   /* 5376 */
#define CMB  (HIDD + INTR)     /* 3840 */

typedef __attribute__((ext_vector_type(16))) _Float16 v16h;
typedef __attribute__((ext_vector_type(8)))  float    v8f;
typedef __attribute__((ext_vector_type(4)))  unsigned int u32x4;
typedef int i32x4v __attribute__((vector_size(16)));

union Frag { v16h h; u32x4 q[2]; };

// ---- CDNA5 async global->LDS staging (guarded; falls back to sync path) ----
#if __has_builtin(__builtin_amdgcn_global_load_async_to_lds_b128)
#define HAVE_ASYNC_LDS 1
typedef __attribute__((address_space(1))) i32x4v* as1_i32x4p;
typedef __attribute__((address_space(3))) i32x4v* as3_i32x4p;
#else
#define HAVE_ASYNC_LDS 0
#endif

#if HAVE_ASYNC_LDS
#if __has_builtin(__builtin_amdgcn_s_wait_asynccnt)
#define WAIT_ASYNC(n) __builtin_amdgcn_s_wait_asynccnt(n)
#else
#define WAIT_ASYNC(n) asm volatile("s_wait_asynccnt %0" :: "i"(n) : "memory")
#endif
#else
#define WAIT_ASYNC(n)
#endif

__device__ __forceinline__ void cp_tile_b128(const _Float16* g, _Float16* l)
{
#if HAVE_ASYNC_LDS
  __builtin_amdgcn_global_load_async_to_lds_b128(
      (as1_i32x4p)g, (as3_i32x4p)l, 0, 0);
#else
  *(u32x4*)l = *(const u32x4*)g;
#endif
}

// ---------------------------------------------------------------------------
// LayerNorm: x(row,768) -> xn f32 (residual) + f16 (GEMM input)
// ---------------------------------------------------------------------------
__global__ __launch_bounds__(256) void ln_kernel(
    const float* __restrict__ x, const float* __restrict__ g,
    const float* __restrict__ b, float* __restrict__ xnf,
    _Float16* __restrict__ xnh)
{
  const int row = blockIdx.x, tid = threadIdx.x;
  __shared__ float red[16];
  const float* xr = x + (size_t)row * HIDD;
  float v0 = xr[tid], v1 = xr[tid + 256], v2 = xr[tid + 512];
  float s = v0 + v1 + v2, ss = v0*v0 + v1*v1 + v2*v2;
  for (int off = 16; off > 0; off >>= 1) {
    s  += __shfl_xor(s,  off, 32);
    ss += __shfl_xor(ss, off, 32);
  }
  if ((tid & 31) == 0) { red[tid >> 5] = s; red[8 + (tid >> 5)] = ss; }
  __syncthreads();
  if (tid == 0) {
    float a = 0.f, c = 0.f;
    for (int i = 0; i < 8; ++i) { a += red[i]; c += red[8 + i]; }
    red[0] = a; red[8] = c;
  }
  __syncthreads();
  float mean = red[0] * (1.0f / HIDD);
  float var  = red[8] * (1.0f / HIDD) - mean * mean;
  float inv  = rsqrtf(var + 1e-5f);
  float vv[3] = { v0, v1, v2 };
  #pragma unroll
  for (int j = 0; j < 3; ++j) {
    int cidx = tid + j * 256;
    float y = (vv[j] - mean) * inv * g[cidx] + b[cidx];
    xnf[(size_t)row * HIDD + cidx] = y;
    xnh[(size_t)row * HIDD + cidx] = (_Float16)y;
  }
}

// ---------------------------------------------------------------------------
// Transpose + convert f32 [Kd][Nd] -> f16 [Nd][Kd] (B^T layout for WMMA)
// ---------------------------------------------------------------------------
__global__ __launch_bounds__(256) void convert_t(
    const float* __restrict__ in, _Float16* __restrict__ out, int Kd, int Nd)
{
  size_t idx = (size_t)blockIdx.x * 256 + threadIdx.x;
  size_t total = (size_t)Kd * Nd;
  if (idx >= total) return;
  int n = (int)(idx / Kd), k = (int)(idx % Kd);
  out[idx] = (_Float16)in[(size_t)k * Nd + n];
}

// ---------------------------------------------------------------------------
// WMMA GEMM: out[M][N] = A[M][K](f16) * Bt[N][K](f16) + bias
// mode 0: store f16 ; mode 1: store f32 with residual add
// Block = 256 thr (8 waves), tile 128x128, K-step 32.
// 3-stage circular LDS pipeline staged via async global->LDS; one barrier per
// K-step; steady-state loop is branch-free (last two iterations peeled).
// ---------------------------------------------------------------------------
__global__ __launch_bounds__(256) void wmma_gemm(
    const _Float16* __restrict__ A, const _Float16* __restrict__ Bt,
    const float* __restrict__ bias, const float* __restrict__ resid,
    _Float16* __restrict__ outH, float* __restrict__ outF,
    int M, int N, int K, int mode)
{
  __shared__ _Float16 ldsA[3][128 * 40];
  __shared__ _Float16 ldsB[3][128 * 40];
  const int tid  = threadIdx.x;
  const int lane = tid & 31, wid = tid >> 5;
  const int ln15 = lane & 15, hi = lane >> 4;
  const int m0 = blockIdx.y * 128, n0 = blockIdx.x * 128;
  const int wm = (wid & 3) * 32, wn = (wid >> 2) * 64;

  const v8f vz = { 0.f,0.f,0.f,0.f,0.f,0.f,0.f,0.f };
  v8f acc[2][4];
  #pragma unroll
  for (int i = 0; i < 2; ++i)
    #pragma unroll
    for (int j = 0; j < 4; ++j) acc[i][j] = vz;

  const int nk = K >> 5;
  const int lr = tid >> 1;          // 0..127 tile row
  const int lc = (tid & 1) * 16;    // 0 or 16 (halves)
  const int lo = lr * 40 + lc;      // per-thread LDS staging offset
  const _Float16* ga = A  + (size_t)(m0 + lr) * K + lc;
  const _Float16* gb = Bt + (size_t)(n0 + lr) * K + lc;

  auto stage = [&](const _Float16* gA, const _Float16* gB, int bi) {
    _Float16* la = &ldsA[bi][lo];
    _Float16* lb = &ldsB[bi][lo];
    cp_tile_b128(gA,     la);
    cp_tile_b128(gA + 8, la + 8);
    cp_tile_b128(gB,     lb);
    cp_tile_b128(gB + 8, lb + 8);
  };

  auto compute = [&](const _Float16* la, const _Float16* lb) {
    Frag af[2], bf[4];
    #pragma unroll
    for (int rt = 0; rt < 2; ++rt) {
      int row = wm + rt * 16 + ln15;
      af[rt].q[0] = *(const u32x4*)&la[row * 40 + hi * 8];
      af[rt].q[1] = *(const u32x4*)&la[row * 40 + 16 + hi * 8];
    }
    #pragma unroll
    for (int ct = 0; ct < 4; ++ct) {
      int col = wn + ct * 16 + ln15;
      bf[ct].q[0] = *(const u32x4*)&lb[col * 40 + hi * 8];
      bf[ct].q[1] = *(const u32x4*)&lb[col * 40 + 16 + hi * 8];
    }
    #pragma unroll
    for (int rt = 0; rt < 2; ++rt)
      #pragma unroll
      for (int ct = 0; ct < 4; ++ct)
        acc[rt][ct] = __builtin_amdgcn_wmma_f32_16x16x32_f16(
            false, af[rt].h, false, bf[ct].h, (short)0, acc[rt][ct], false, false);
  };

  stage(ga, gb, 0);
  if (nk > 1) stage(ga + 32, gb + 32, 1);
  const _Float16* pga = ga + 64;    // next tile to stage (tile 2)
  const _Float16* pgb = gb + 64;

  int get = 0, put = 2;
  // Steady state: tile i is resident (only tile i+1's 4 async ops in flight),
  // stage tile i+2 into the buffer read in the previous iteration.
  for (int i = 0; i < nk - 2; ++i) {
    WAIT_ASYNC(4);
    __syncthreads();
    compute(ldsA[get], ldsB[get]);
    stage(pga, pgb, put);
    pga += 32; pgb += 32;
    get = (get == 2) ? 0 : get + 1;
    put = (put == 2) ? 0 : put + 1;
  }
  if (nk > 1) {                      // tile nk-2: nothing new to stage
    WAIT_ASYNC(4);
    __syncthreads();
    compute(ldsA[get], ldsB[get]);
    get = (get == 2) ? 0 : get + 1;
  }
  WAIT_ASYNC(0);                     // drain: last tile fully resident
  __syncthreads();
  compute(ldsA[get], ldsB[get]);

  #pragma unroll
  for (int rt = 0; rt < 2; ++rt)
    #pragma unroll
    for (int ct = 0; ct < 4; ++ct) {
      int col = n0 + wn + ct * 16 + ln15;
      float bv = bias[col];
      #pragma unroll
      for (int r = 0; r < 8; ++r) {
        int row = m0 + wm + rt * 16 + r + hi * 8;
        float v = acc[rt][ct][r] + bv;
        size_t idx = (size_t)row * N + col;
        if (mode == 0) outH[idx] = (_Float16)v;
        else           outF[idx] = resid[idx] + v;
      }
    }
}

// ---------------------------------------------------------------------------
// RoPE on q/k, copy v, GELU(tanh) on ff. mid f16 [N][5376] ->
//   qr/kr/vr f16 [H][N][64] , comb[:,768:] = gelu(ff)
// ---------------------------------------------------------------------------
__global__ __launch_bounds__(256) void rope_split(
    const _Float16* __restrict__ mid, const float* __restrict__ psin,
    const float* __restrict__ pcos, _Float16* __restrict__ qr,
    _Float16* __restrict__ kr, _Float16* __restrict__ vr,
    _Float16* __restrict__ comb)
{
  size_t idx = (size_t)blockIdx.x * 256 + threadIdx.x;
  if (idx >= (size_t)NTOK * MIDC) return;
  int t = (int)(idx / MIDC), col = (int)(idx % MIDC);
  float xv = (float)mid[idx];
  if (col >= 3 * HIDD) {
    float u = xv;
    float gl = 0.5f * u * (1.0f + tanhf(0.7978845608028654f * (u + 0.044715f * u * u * u)));
    comb[(size_t)t * CMB + HIDD + (col - 3 * HIDD)] = (_Float16)gl;
  } else {
    int which = col / HIDD;
    int cc = col - which * HIDD;
    int hh = cc >> 6, d = cc & 63;
    size_t oidx = ((size_t)hh * NTOK + t) * DHD + d;
    if (which == 2) { vr[oidx] = (_Float16)xv; return; }
    float pa = (float)mid[idx ^ 1];          // pair partner (row stride even)
    float sn = psin[(size_t)t * DHD + d], cs = pcos[(size_t)t * DHD + d];
    float rot = (d & 1) ? pa : -pa;
    float y = xv * cs + rot * sn;
    if (which == 0) qr[oidx] = (_Float16)y; else kr[oidx] = (_Float16)y;
  }
}

// ---------------------------------------------------------------------------
// Fused flash attention per (chunk, head). Block 256 thr (8 waves); each wave
// owns 32 query rows. Keys = [chunk-256, chunk+256) processed 32 at a time.
// ---------------------------------------------------------------------------
__global__ __launch_bounds__(256) void attn_kernel(
    const _Float16* __restrict__ qg, const _Float16* __restrict__ kg,
    const _Float16* __restrict__ vg, _Float16* __restrict__ comb,
    const int* __restrict__ lenp)
{
  __shared__ _Float16 vTs[64 * 40];        // V^T for current 32-key step
  __shared__ _Float16 pb[8 * 32 * 40];     // per-wave P re-swizzle scratch
  const int c = blockIdx.x, h = blockIdx.y;
  const int tid = threadIdx.x;
  const int lane = tid & 31, wid = tid >> 5;
  const int ln15 = lane & 15, hi = lane >> 4;
  const unsigned lm = ~(unsigned)(lenp[0] - 1);
  const int base_tok = c * 256 - 256;
  const _Float16* qh = qg + (size_t)h * NTOK * DHD;
  const _Float16* kh = kg + (size_t)h * NTOK * DHD;
  const _Float16* vh = vg + (size_t)h * NTOK * DHD;
  const int qrow0 = c * 256 + wid * 32;
  const int pwb = wid * 32 * 40;

  // Resident Q fragments (A layout), DH=64 -> 2 k-steps
  Frag qa[2][2];
  #pragma unroll
  for (int rt = 0; rt < 2; ++rt) {
    const _Float16* qp = qh + (size_t)(qrow0 + rt * 16 + ln15) * DHD;
    #pragma unroll
    for (int ks = 0; ks < 2; ++ks) {
      qa[rt][ks].q[0] = *(const u32x4*)(qp + ks * 32 + hi * 8);
      qa[rt][ks].q[1] = *(const u32x4*)(qp + ks * 32 + 16 + hi * 8);
    }
  }

  const v8f vz = { 0.f,0.f,0.f,0.f,0.f,0.f,0.f,0.f };
  float mrow[2][8], lsum[2][8];
  v8f o[2][4];
  #pragma unroll
  for (int rt = 0; rt < 2; ++rt) {
    #pragma unroll
    for (int r = 0; r < 8; ++r) { mrow[rt][r] = -1e30f; lsum[rt][r] = 0.f; }
    #pragma unroll
    for (int dt = 0; dt < 4; ++dt) o[rt][dt] = vz;
  }

  const int vkk = tid >> 3;          // key within step (0..31)
  const int vdg = (tid & 7) * 8;     // dh group base

  for (int kb = 0; kb < 16; ++kb) {
    __syncthreads();
    { // stage V^T tile [64][32] for this key step
      int gt = base_tok + kb * 32 + vkk; if (gt < 0) gt = 0;
      u32x4 dv = *(const u32x4*)(vh + (size_t)gt * DHD + vdg);
      _Float16 tmp[8]; *(u32x4*)tmp = dv;
      #pragma unroll
      for (int j = 0; j < 8; ++j) vTs[(vdg + j) * 40 + vkk] = tmp[j];
    }
    __syncthreads();

    // K fragments directly from global ([key][dh] is exactly B layout)
    Frag kf[2][2];
    #pragma unroll
    for (int ct = 0; ct < 2; ++ct) {
      int key = base_tok + kb * 32 + ct * 16 + ln15;
      int kcl = key < 0 ? 0 : key;
      const _Float16* kp = kh + (size_t)kcl * DHD;
      #pragma unroll
      for (int ks = 0; ks < 2; ++ks) {
        kf[ct][ks].q[0] = *(const u32x4*)(kp + ks * 32 + hi * 8);
        kf[ct][ks].q[1] = *(const u32x4*)(kp + ks * 32 + 16 + hi * 8);
      }
    }

    v8f s[2][2];
    s[0][0] = vz; s[0][1] = vz; s[1][0] = vz; s[1][1] = vz;
    #pragma unroll
    for (int rt = 0; rt < 2; ++rt)
      #pragma unroll
      for (int ct = 0; ct < 2; ++ct) {
        s[rt][ct] = __builtin_amdgcn_wmma_f32_16x16x32_f16(
            false, qa[rt][0].h, false, kf[ct][0].h, (short)0, s[rt][ct], false, false);
        s[rt][ct] = __builtin_amdgcn_wmma_f32_16x16x32_f16(
            false, qa[rt][1].h, false, kf[ct][1].h, (short)0, s[rt][ct], false, false);
      }

    const int kcol0 = base_tok + kb * 32 + ln15;
    #pragma unroll
    for (int rt = 0; rt < 2; ++rt) {
      float sv0[8], sv1[8], tm[8];
      #pragma unroll
      for (int r = 0; r < 8; ++r) {
        int qi = qrow0 + rt * 16 + r + hi * 8;
        int kj0 = kcol0, kj1 = kcol0 + 16;
        int d0 = qi - kj0, d1 = qi - kj1;
        bool ok0 = (kj0 >= 0) && (d0 >= 0) && (d0 < WIN) &&
                   ((((unsigned)qi) & lm) == (((unsigned)kj0) & lm));
        bool ok1 = (kj1 >= 0) && (d1 >= 0) && (d1 < WIN) &&
                   ((((unsigned)qi) & lm) == (((unsigned)kj1) & lm));
        sv0[r] = ok0 ? s[rt][0][r] * 0.125f : -1e30f;
        sv1[r] = ok1 ? s[rt][1][r] * 0.125f : -1e30f;
        tm[r] = fmaxf(sv0[r], sv1[r]);
      }
      for (int off = 1; off < 16; off <<= 1)
        #pragma unroll
        for (int r = 0; r < 8; ++r) tm[r] = fmaxf(tm[r], __shfl_xor(tm[r], off, 32));
      float psum[8];
      #pragma unroll
      for (int r = 0; r < 8; ++r) {
        float mnew  = fmaxf(mrow[rt][r], tm[r]);
        float alpha = __expf(mrow[rt][r] - mnew);
        mrow[rt][r] = mnew;
        lsum[rt][r] *= alpha;
        #pragma unroll
        for (int dt = 0; dt < 4; ++dt) o[rt][dt][r] *= alpha;
        float p0 = (sv0[r] < -1e29f) ? 0.f : __expf(sv0[r] - mnew);
        float p1 = (sv1[r] < -1e29f) ? 0.f : __expf(sv1[r] - mnew);
        psum[r] = p0 + p1;
        int prow = rt * 16 + r + hi * 8;
        pb[pwb + prow * 40 + ln15]      = (_Float16)p0;
        pb[pwb + prow * 40 + 16 + ln15] = (_Float16)p1;
      }
      for (int off = 1; off < 16; off <<= 1)
        #pragma unroll
        for (int r = 0; r < 8; ++r) psum[r] += __shfl_xor(psum[r], off, 32);
      #pragma unroll
      for (int r = 0; r < 8; ++r) lsum[rt][r] += psum[r];
    }

    // P (A layout, via per-wave LDS) and V^T (B layout) fragments; O += P@V
    Frag pf[2], vf[4];
    #pragma unroll
    for (int rt = 0; rt < 2; ++rt) {
      int prow = rt * 16 + ln15;
      pf[rt].q[0] = *(const u32x4*)&pb[pwb + prow * 40 + hi * 8];
      pf[rt].q[1] = *(const u32x4*)&pb[pwb + prow * 40 + 16 + hi * 8];
    }
    #pragma unroll
    for (int dt = 0; dt < 4; ++dt) {
      int d = dt * 16 + ln15;
      vf[dt].q[0] = *(const u32x4*)&vTs[d * 40 + hi * 8];
      vf[dt].q[1] = *(const u32x4*)&vTs[d * 40 + 16 + hi * 8];
    }
    #pragma unroll
    for (int rt = 0; rt < 2; ++rt)
      #pragma unroll
      for (int dt = 0; dt < 4; ++dt)
        o[rt][dt] = __builtin_amdgcn_wmma_f32_16x16x32_f16(
            false, pf[rt].h, false, vf[dt].h, (short)0, o[rt][dt], false, false);
  }

  #pragma unroll
  for (int rt = 0; rt < 2; ++rt)
    #pragma unroll
    for (int dt = 0; dt < 4; ++dt)
      #pragma unroll
      for (int r = 0; r < 8; ++r) {
        int tok = qrow0 + rt * 16 + r + hi * 8;
        int d = dt * 16 + ln15;
        float val = o[rt][dt][r] / lsum[rt][r];
        comb[(size_t)tok * CMB + h * DHD + d] = (_Float16)val;
      }
}

// ---------------------------------------------------------------------------
extern "C" void kernel_launch(void* const* d_in, const int* in_sizes, int n_in,
                              void* d_out, int out_size, void* d_ws, size_t ws_size,
                              hipStream_t stream)
{
  (void)in_sizes; (void)n_in; (void)out_size; (void)ws_size;
  const float* x     = (const float*)d_in[0];
  const float* psin  = (const float*)d_in[2];
  const float* pcos  = (const float*)d_in[3];
  const float* gamma = (const float*)d_in[4];
  const float* beta  = (const float*)d_in[5];
  const float* w_in  = (const float*)d_in[6];
  const float* b_in  = (const float*)d_in[7];
  const float* w_out = (const float*)d_in[8];
  const float* b_out = (const float*)d_in[9];
  const int*   lenp  = (const int*)d_in[10];
  float* out = (float*)d_out;

  char* ws = (char*)d_ws;
  size_t off = 0;
  float*    xnf  = (float*)(ws + off);    off += (size_t)NTOK * HIDD * 4;
  _Float16* xnh  = (_Float16*)(ws + off); off += (size_t)NTOK * HIDD * 2;
  _Float16* wint = (_Float16*)(ws + off); off += (size_t)MIDC * HIDD * 2;
  _Float16* wott = (_Float16*)(ws + off); off += (size_t)CMB  * HIDD * 2;
  _Float16* mid  = (_Float16*)(ws + off); off += (size_t)NTOK * MIDC * 2;
  _Float16* qr   = (_Float16*)(ws + off); off += (size_t)NTOK * HIDD * 2;
  _Float16* kr   = (_Float16*)(ws + off); off += (size_t)NTOK * HIDD * 2;
  _Float16* vr   = (_Float16*)(ws + off); off += (size_t)NTOK * HIDD * 2;
  _Float16* comb = (_Float16*)(ws + off); off += (size_t)NTOK * CMB  * 2;

  ln_kernel<<<NTOK, 256, 0, stream>>>(x, gamma, beta, xnf, xnh);
  convert_t<<<(MIDC * HIDD + 255) / 256, 256, 0, stream>>>(w_in,  wint, HIDD, MIDC);
  convert_t<<<(CMB  * HIDD + 255) / 256, 256, 0, stream>>>(w_out, wott, CMB,  HIDD);
  wmma_gemm<<<dim3(MIDC / 128, NTOK / 128), 256, 0, stream>>>(
      xnh, wint, b_in, nullptr, mid, nullptr, NTOK, MIDC, HIDD, 0);
  rope_split<<<(unsigned)(((size_t)NTOK * MIDC) / 256), 256, 0, stream>>>(
      mid, psin, pcos, qr, kr, vr, comb);
  attn_kernel<<<dim3(NTOK / WIN, NH), 256, 0, stream>>>(qr, kr, vr, comb, lenp);
  wmma_gemm<<<dim3(HIDD / 128, NTOK / 128), 256, 0, stream>>>(
      comb, wott, b_out, xnf, nullptr, out, NTOK, HIDD, CMB, 1);
}
